// Gtu2d_76940044140639
// MI455X (gfx1250) — matrix operand order repaired and checked by reading, not compiled, not run on hardware
//
#include <hip/hip_runtime.h>

typedef __attribute__((ext_vector_type(16))) __bf16 v16bf;
typedef __attribute__((ext_vector_type(8)))  float  v8f;

// ---------- bf16 helpers (stored as ushort) ----------
__device__ __forceinline__ unsigned short f2bf(float f) {
    union { float f; unsigned u; } x; x.f = f;
    unsigned r = x.u + 0x7FFFu + ((x.u >> 16) & 1u);   // round-to-nearest-even
    return (unsigned short)(r >> 16);
}
__device__ __forceinline__ float bf2f(unsigned short s) {
    union { unsigned u; float f; } x; x.u = ((unsigned)s) << 16; return x.f;
}
// fast transcendental paths (single v_rcp_f32 / v_rsq_f32 trans ops)
__device__ __forceinline__ float fast_rcp(float x)  { return __builtin_amdgcn_rcpf(x); }
__device__ __forceinline__ float fast_rsqrt(float x){ return __builtin_amdgcn_rsqf(x); }

#define WMMA_BF16(A, B, C) \
    __builtin_amdgcn_wmma_f32_16x16x32_bf16(false, (A), false, (B), (short)0, (C), false, false)

// ---------------------------------------------------------------------------
// f32 -> bf16 convert
// ---------------------------------------------------------------------------
__global__ void cvt_kernel(const float* __restrict__ in,
                           unsigned short* __restrict__ out, int n) {
    int i = blockIdx.x * 256 + threadIdx.x;
    if (i < n) out[i] = f2bf(in[i]);
}

// ---------------------------------------------------------------------------
// RPE MLP: one thread per relative position k=0..255 -> a[k][384]
// ---------------------------------------------------------------------------
struct RpeParams {
    const float *pos_W, *pos_b;
    const float *lg[3], *lbe[3], *lW[3], *lb[3];
    const float *out_g, *out_be, *out_W, *out_b;
};

__global__ void rpe_kernel(RpeParams p1, RpeParams p2,
                           float* __restrict__ a1, float* __restrict__ a2) {
    const RpeParams p = (blockIdx.x == 0) ? p1 : p2;
    float* out = (blockIdx.x == 0) ? a1 : a2;
    int k = threadIdx.x;                    // 0..255
    float rel;
    if (k == 0 || k == 128) rel = 0.f;
    else if (k < 128)       rel = (float)k;
    else                    rel = (float)(k - 256);

    float h[64], z[64];
    for (int j = 0; j < 64; ++j) h[j] = rel * p.pos_W[j] + p.pos_b[j];

    for (int L = 0; L < 3; ++L) {
        float mu = 0.f;  for (int j = 0; j < 64; ++j) mu += h[j];
        mu *= (1.f / 64.f);
        float var = 0.f; for (int j = 0; j < 64; ++j) { float d = h[j] - mu; var += d * d; }
        var *= (1.f / 64.f);
        float inv = fast_rsqrt(var + 1e-5f);
        for (int j = 0; j < 64; ++j) {
            float t = (h[j] - mu) * inv * p.lg[L][j] + p.lbe[L][j];
            z[j] = t > 0.f ? t : 0.f;
        }
        for (int j = 0; j < 64; ++j) {
            float acc = p.lb[L][j];
            const float* w = p.lW[L] + j * 64;
            for (int i = 0; i < 64; ++i) acc += z[i] * w[i];
            h[j] = acc;
        }
    }
    // final LN+relu then 384x64 projection
    float mu = 0.f;  for (int j = 0; j < 64; ++j) mu += h[j];
    mu *= (1.f / 64.f);
    float var = 0.f; for (int j = 0; j < 64; ++j) { float d = h[j] - mu; var += d * d; }
    var *= (1.f / 64.f);
    float inv = fast_rsqrt(var + 1e-5f);
    for (int j = 0; j < 64; ++j) {
        float t = (h[j] - mu) * inv * p.out_g[j] + p.out_be[j];
        z[j] = t > 0.f ? t : 0.f;
    }
    for (int j = 0; j < 384; ++j) {
        float acc = p.out_b[j];
        const float* w = p.out_W + j * 64;
        for (int i = 0; i < 64; ++i) acc += z[i] * w[i];
        out[k * 384 + j] = acc;
    }
}

// ---------------------------------------------------------------------------
// u/v projection: C[row][hd] = silu(xb[row][:] @ W[hd][:] + b[hd])
// stored hd-major: Out[hd*65536 + row] (bf16).  8 waves x 16-row blocks.
// ---------------------------------------------------------------------------
__global__ __launch_bounds__(256) void uv_gemm(
    const unsigned short* __restrict__ xb,
    const unsigned short* __restrict__ uwb,
    const unsigned short* __restrict__ vwb,
    const float* __restrict__ u_b, const float* __restrict__ v_b,
    unsigned short* __restrict__ U, unsigned short* __restrict__ V0) {

    int l = threadIdx.x & 31, wave = threadIdx.x >> 5;
    int lm = l & 15, lh = l >> 4;
    int m0 = blockIdx.x * 128 + wave * 16;

    v16bf a[4];                                        // A frags: x rows, reused 48x
    for (int ki = 0; ki < 4; ++ki) {
        const unsigned short* pa = xb + (unsigned)(m0 + lm) * 128u + ki * 32 + lh * 8;
        union { uint4 u4[2]; v16bf v; } t;
        t.u4[0] = *(const uint4*)pa;
        t.u4[1] = *(const uint4*)(pa + 16);
        a[ki] = t.v;
    }
    for (int j = 0; j < 48; ++j) {
        const unsigned short* Wb = (j < 24) ? uwb : vwb;
        const float* bias        = (j < 24) ? u_b : v_b;
        unsigned short* Out      = (j < 24) ? U : V0;
        int n0 = (j % 24) * 16;
        v8f c = {};
        for (int ki = 0; ki < 4; ++ki) {
            const unsigned short* pb = Wb + (unsigned)(n0 + lm) * 128u + ki * 32 + lh * 8;
            union { uint4 u4[2]; v16bf v; } t;
            t.u4[0] = *(const uint4*)pb;
            t.u4[1] = *(const uint4*)(pb + 16);
            c = WMMA_BF16(a[ki], t.v, c);
        }
        float bv = bias[n0 + lm];
        union { unsigned short s[8]; uint4 u4; } st;
        for (int r = 0; r < 8; ++r) {
            float s = c[r] + bv;
            float sig = fast_rcp(1.f + __expf(-s));    // silu via single v_rcp_f32
            st.s[r] = f2bf(s * sig);
        }
        *(uint4*)(Out + (unsigned)(n0 + lm) * 65536u + m0 + lh * 8) = st.u4;
    }
}

// ---------------------------------------------------------------------------
// Toeplitz GEMM along one axis for one (hd, batch):
//   D[128 x 128cols] = T^{hd}[128x128] @ B,  T[i][j] = coeff[(i-j)&255]
// strided==0 : In element (K=k, N=n) at base + n*128 + k   (contiguous K)
// strided==1 : In element (K=k, N=n) at base + k*128 + n   (LDS transpose)
// Output always stored direct: Out[base + col*128 + M].
// ---------------------------------------------------------------------------
__global__ __launch_bounds__(256) void toeplitz_gemm(
    const float* __restrict__ aTab,
    const unsigned short* __restrict__ In,
    unsigned short* __restrict__ Out, int strided) {

    __shared__ __align__(16) unsigned short c_lds[256];
    __shared__ __align__(16) unsigned short Bt[16][128];

    int tid = threadIdx.x;
    int hd = blockIdx.x, bb = blockIdx.y;
    c_lds[tid] = f2bf(aTab[tid * 384 + hd]);
    __syncthreads();

    int l = tid & 31, wave = tid >> 5;
    int lm = l & 15, lh = l >> 4;
    int m0 = wave * 16;

    v16bf a[4];                                        // Toeplitz A frags, built once
    for (int ki = 0; ki < 4; ++ki) {
        union { unsigned short s[16]; v16bf v; } t;
        int kb = ki * 32 + lh * 8;
        for (int j = 0; j < 8; ++j) {
            t.s[j]     = c_lds[(m0 + lm - (kb + j)) & 255];
            t.s[8 + j] = c_lds[(m0 + lm - (kb + 16 + j)) & 255];
        }
        a[ki] = t.v;
    }

    const unsigned base = (unsigned)hd * 65536u + (unsigned)bb * 16384u;
    for (int nb = 0; nb < 8; ++nb) {
        if (!strided) {
            int n = tid >> 4, ch = tid & 15;
            const unsigned short* src = In + base + (unsigned)(nb * 16 + n) * 128u + ch * 8;
            if (nb < 7) __builtin_prefetch(src + 16 * 128, 0, 1);
            *(uint4*)&Bt[n][ch * 8] = *(const uint4*)src;
        } else {
            int k = tid >> 1, h8 = (tid & 1) * 8;
            union { uint4 u4; unsigned short s[8]; } v;
            const unsigned short* src = In + base + (unsigned)k * 128u + nb * 16 + h8;
            if (nb < 7) __builtin_prefetch(src + 16, 0, 1);
            v.u4 = *(const uint4*)src;
            for (int jj = 0; jj < 8; ++jj) Bt[h8 + jj][k] = v.s[jj];
        }
        __syncthreads();

        v16bf b[4];
        for (int ki = 0; ki < 4; ++ki) {
            const unsigned short* pb = &Bt[lm][ki * 32 + lh * 8];
            union { uint4 u4[2]; v16bf v; } t;
            t.u4[0] = *(const uint4*)pb;
            t.u4[1] = *(const uint4*)(pb + 16);
            b[ki] = t.v;
        }
        v8f c = {};
        for (int ki = 0; ki < 4; ++ki) c = WMMA_BF16(a[ki], b[ki], c);

        union { unsigned short s[8]; uint4 u4; } st;
        for (int r = 0; r < 8; ++r) st.s[r] = f2bf(c[r]);
        *(uint4*)(Out + base + (unsigned)(nb * 16 + lm) * 128u + m0 + lh * 8) = st.u4;
        __syncthreads();
    }
}

// ---------------------------------------------------------------------------
// gating: g[hd][row] = u[hd][row] * (o1[hd][b][w][y] + o2[hd][b][y][w])
// ---------------------------------------------------------------------------
__global__ void gate_kernel(const unsigned short* __restrict__ U,
                            const unsigned short* __restrict__ O1,
                            const unsigned short* __restrict__ O2,
                            unsigned short* __restrict__ G) {
    unsigned idx = blockIdx.x * 256u + threadIdx.x;        // < 384*65536
    unsigned hd = idx >> 16, p = idx & 65535u;
    unsigned b = p >> 14, q = p & 16383u;
    unsigned y = q >> 7, w = q & 127u;
    float o1 = bf2f(O1[(hd << 16) + (b << 14) + (w << 7) + y]);
    float o2 = bf2f(O2[idx]);
    float u  = bf2f(U[idx]);
    G[idx] = f2bf(u * (o1 + o2));
}

// ---------------------------------------------------------------------------
// output projection: out[row][dm] = g[row][:] @ o_W[dm][:] + o_b[dm]
// A staged through LDS (transpose from hd-major g); 8 waves split N=128.
// ---------------------------------------------------------------------------
__global__ __launch_bounds__(256) void out_gemm(
    const unsigned short* __restrict__ G,
    const unsigned short* __restrict__ owb,
    const float* __restrict__ o_b, float* __restrict__ out) {

    __shared__ __align__(16) unsigned short At[16][384];
    int tid = threadIdx.x;
    int m0 = blockIdx.x * 16;

    for (int s = tid; s < 768; s += 256) {               // At[r][hd] = G[hd][m0+r]
        int hd = s >> 1, half = (s & 1) * 8;
        union { uint4 u4; unsigned short e[8]; } v;
        v.u4 = *(const uint4*)(G + (unsigned)hd * 65536u + m0 + half);
        for (int r = 0; r < 8; ++r) At[half + r][hd] = v.e[r];
    }
    __syncthreads();

    int l = tid & 31, wave = tid >> 5;
    int lm = l & 15, lh = l >> 4;
    int n0 = wave * 16;

    v8f c = {};
    for (int ki = 0; ki < 12; ++ki) {
        union { uint4 u4[2]; v16bf v; } ta, tb;
        const unsigned short* pa = &At[lm][ki * 32 + lh * 8];
        ta.u4[0] = *(const uint4*)pa;
        ta.u4[1] = *(const uint4*)(pa + 16);
        const unsigned short* pb = owb + (unsigned)(n0 + lm) * 384u + ki * 32 + lh * 8;
        tb.u4[0] = *(const uint4*)pb;
        tb.u4[1] = *(const uint4*)(pb + 16);
        c = WMMA_BF16(ta.v, tb.v, c);
    }
    float bv = o_b[n0 + lm];
    for (int r = 0; r < 8; ++r)
        out[(unsigned)(m0 + lh * 8 + r) * 128u + n0 + lm] = c[r] + bv;
}

// ---------------------------------------------------------------------------
// host side
// ---------------------------------------------------------------------------
static RpeParams mk_params(void* const* p) {
    RpeParams r;
    r.pos_W = (const float*)p[0];  r.pos_b  = (const float*)p[1];
    for (int i = 0; i < 3; ++i) {
        r.lg[i]  = (const float*)p[2 + i * 4 + 0];
        r.lbe[i] = (const float*)p[2 + i * 4 + 1];
        r.lW[i]  = (const float*)p[2 + i * 4 + 2];
        r.lb[i]  = (const float*)p[2 + i * 4 + 3];
    }
    r.out_g = (const float*)p[14]; r.out_be = (const float*)p[15];
    r.out_W = (const float*)p[16]; r.out_b  = (const float*)p[17];
    return r;
}

extern "C" void kernel_launch(void* const* d_in, const int* in_sizes, int n_in,
                              void* d_out, int out_size, void* d_ws, size_t ws_size,
                              hipStream_t stream) {
    const float* x   = (const float*)d_in[0];
    const float* u_W = (const float*)d_in[1];
    const float* u_b = (const float*)d_in[2];
    const float* v_W = (const float*)d_in[3];
    const float* v_b = (const float*)d_in[4];
    const float* o_W = (const float*)d_in[5];
    const float* o_b = (const float*)d_in[6];
    RpeParams t1 = mk_params(d_in + 7);
    RpeParams t2 = mk_params(d_in + 25);

    char* ws = (char*)d_ws;
    // workspace layout (bytes)
    float*          a1  = (float*)(ws + 0);                       //  256*384 f32
    float*          a2  = (float*)(ws + 393216);
    unsigned short* xb  = (unsigned short*)(ws + 786432);         //  8388608 bf16
    unsigned short* uwb = (unsigned short*)(ws + 17563648);       //  49152 bf16
    unsigned short* vwb = (unsigned short*)(ws + 17661952);
    unsigned short* owb = (unsigned short*)(ws + 17760256);
    unsigned short* U   = (unsigned short*)(ws + 17858560);       //  384*65536 bf16
    unsigned short* V0  = (unsigned short*)(ws + 68190208);
    unsigned short* W1  = (unsigned short*)(ws + 118521856);      //  X1 / Y / G
    unsigned short* W2  = (unsigned short*)(ws + 168853504);      //  O1
    unsigned short* O2  = V0;                                     //  reuse (V0 dead after pass 3)

    cvt_kernel<<<(8388608 + 255) / 256, 256, 0, stream>>>(x, xb, 8388608);
    cvt_kernel<<<(49152 + 255) / 256, 256, 0, stream>>>(u_W, uwb, 49152);
    cvt_kernel<<<(49152 + 255) / 256, 256, 0, stream>>>(v_W, vwb, 49152);
    cvt_kernel<<<(49152 + 255) / 256, 256, 0, stream>>>(o_W, owb, 49152);

    rpe_kernel<<<2, 256, 0, stream>>>(t1, t2, a1, a2);

    uv_gemm<<<512, 256, 0, stream>>>(xb, uwb, vwb, u_b, v_b, U, V0);

    // o1 = T2_H( T1_W(v) );  o2 = T1_W( T2_H(v) )
    toeplitz_gemm<<<dim3(384, 4), 256, 0, stream>>>(a1, V0, W1, 0);   // X1 = T1 along w
    toeplitz_gemm<<<dim3(384, 4), 256, 0, stream>>>(a2, W1, W2, 1);   // O1 = T2 along y
    toeplitz_gemm<<<dim3(384, 4), 256, 0, stream>>>(a2, V0, W1, 1);   // Y  = T2 along y
    toeplitz_gemm<<<dim3(384, 4), 256, 0, stream>>>(a1, W1, O2, 1);   // O2 = T1 along w

    gate_kernel<<<98304, 256, 0, stream>>>(U, W2, O2, W1);            // G = u*(o1+o2)

    out_gemm<<<4096, 256, 0, stream>>>(W1, owb, o_b, (float*)d_out);
}
